// GRFieldManager_70377334112328
// MI455X (gfx1250) — compile-verified, matrix operand order checked by reading, not run on hardware
//
#include <hip/hip_runtime.h>
#include <hip/hip_bf16.h>

typedef _Float16 v16h __attribute__((ext_vector_type(16)));
typedef _Float16 v8h  __attribute__((ext_vector_type(8)));
typedef _Float16 v2h  __attribute__((ext_vector_type(2)));
typedef float    v8f  __attribute__((ext_vector_type(8)));

#define NPIX   (8 * 256 * 256)
#define NELEM  (NPIX * 10)
#define PI_F   3.14159265358979f
#define COUPLING_F 25.132741228718345f   /* 8*pi*G/c^4, G=c=1 */

// ws float layout: [0..2]=softmax weights, [3]=inv_max, [4]=raw max,
// [16 .. 16+NPIX) = potential, [16+NPIX ..) = pre-smooth metric
#define WS_POT_OFF 16

// ---------------------------------------------------------------------------
// 0a. init scratch scalars
__global__ void k_init(float* wsf) {
    if (threadIdx.x == 0 && blockIdx.x == 0) wsf[4] = 0.0f;
}

// 0b. global max of mass (mass >= 0, so uint-bit compare == float compare)
__global__ void k_maxreduce(const float* __restrict__ mass, float* wsf, int n) {
    __shared__ float smax[256];
    int tid = threadIdx.x;
    float m = 0.0f;
    for (int i = blockIdx.x * blockDim.x + tid; i < n; i += gridDim.x * blockDim.x)
        m = fmaxf(m, mass[i]);
    smax[tid] = m;
    __syncthreads();
    for (int s = 128; s > 0; s >>= 1) {
        if (tid < s) smax[tid] = fmaxf(smax[tid], smax[tid + s]);
        __syncthreads();
    }
    if (tid == 0)
        atomicMax((unsigned int*)&wsf[4], __float_as_uint(smax[0]));
}

// 0c. softmax(solver_params) + 1/(max+1e-8)
__global__ void k_finalize(const float* __restrict__ sp, float* wsf) {
    if (threadIdx.x != 0 || blockIdx.x != 0) return;
    float a = sp[0], b = sp[1], c = sp[2];
    float mx = fmaxf(a, fmaxf(b, c));
    float ea = expf(a - mx), eb = expf(b - mx), ec = expf(c - mx);
    float s = ea + eb + ec;
    wsf[0] = ea / s;
    wsf[1] = eb / s;
    wsf[2] = ec / s;
    wsf[3] = 1.0f / (wsf[4] + 1e-8f);
}

// ---------------------------------------------------------------------------
// 1. 7x7 Green's-function potential: pot = conv(mass, -1/(2*pi*r)) * inv_max
__global__ void k_green(const float* __restrict__ mass, const float* __restrict__ wsf,
                        float* __restrict__ pot) {
    __shared__ float tile[22][22];
    __shared__ float gk[49];
    int b  = blockIdx.z;
    int h0 = blockIdx.y * 16;
    int w0 = blockIdx.x * 16;
    int tid = threadIdx.y * 16 + threadIdx.x;
    if (tid < 49) {
        float dy = (float)(tid / 7) - 3.0f;
        float dx = (float)(tid % 7) - 3.0f;
        float r2 = dy * dy + dx * dx;
        gk[tid] = (r2 > 0.0f) ? (-1.0f / (2.0f * PI_F * sqrtf(r2))) : 0.0f;
    }
    const float* mb = mass + (long)b * 65536;
    for (int e = tid; e < 484; e += 256) {
        int ty = e / 22, tx = e % 22;
        int hh = h0 + ty - 3, ww = w0 + tx - 3;
        float v = 0.0f;
        if (hh >= 0 && hh < 256 && ww >= 0 && ww < 256) v = mb[hh * 256 + ww];
        tile[ty][tx] = v;
    }
    __syncthreads();
    float inv = wsf[3];
    float s = 0.0f;
#pragma unroll
    for (int i = 0; i < 7; ++i)
#pragma unroll
        for (int j = 0; j < 7; ++j)
            s = fmaf(gk[i * 7 + j], tile[threadIdx.y + i][threadIdx.x + j], s);
    pot[(long)b * 65536 + (h0 + threadIdx.y) * 256 + (w0 + threadIdx.x)] = s * inv;
}

// ---------------------------------------------------------------------------
// 2. Main kernel: fused MLP stress-energy tensor (WMMA second layers) + metric
//    update + potential correction + signature clamps.
#define WAVES   4
#define ITERS   4
#define ASTRIDE 264   /* halves per A row: 528 B, 16B-aligned, conflict-padded */

__device__ __forceinline__ float gelu_exact(float x) {
    return 0.5f * x * (1.0f + erff(x * 0.70710678118654752f));
}

__global__ __launch_bounds__(128) void k_em(
    const float* __restrict__ mass, const float* __restrict__ vel,
    const float* __restrict__ metric_init,
    const float* __restrict__ e_w1, const float* __restrict__ e_b1,
    const float* __restrict__ e_w2, const float* __restrict__ e_b2,
    const float* __restrict__ m_w1, const float* __restrict__ m_b1,
    const float* __restrict__ m_w2, const float* __restrict__ m_b2,
    const float* __restrict__ s_w1, const float* __restrict__ s_b1,
    const float* __restrict__ s_w2, const float* __restrict__ s_b2,
    const float* __restrict__ wsf, const float* __restrict__ pot,
    float* __restrict__ metric_pre)
{
    // First-layer weights re-packed by combined hidden index j in [0,256):
    // j<64 -> energy net, j<128 -> momentum net, else stress net.
    __shared__ float w1m[256], w1x[256], w1y[256], w1z[256], bb1[256];
    // Combined second-layer weights, pre-swizzled into the per-lane WMMA B
    // fragment layout: fragB[(chunk*32 + lane)*16 + half]
    __shared__ _Float16 fragB[8 * 512];
    // Per-wave A staging: 16 pixel rows x 256 hiddens (f16, padded stride)
    __shared__ _Float16 Abuf[WAVES * 16 * ASTRIDE];

    int tid = threadIdx.x;

    for (int j = tid; j < 256; j += 128) {
        float a0, a1, a2, a3, bv;
        if (j < 64)       { a0 = e_w1[j];       a1 = e_w1[64 + j];   a2 = e_w1[128 + j];  a3 = e_w1[192 + j];  bv = e_b1[j]; }
        else if (j < 128) { int q = j - 64;  a0 = m_w1[q]; a1 = m_w1[64 + q];  a2 = m_w1[128 + q];  a3 = m_w1[192 + q];  bv = m_b1[q]; }
        else              { int q = j - 128; a0 = s_w1[q]; a1 = s_w1[128 + q]; a2 = s_w1[256 + q];  a3 = s_w1[384 + q];  bv = s_b1[q]; }
        w1m[j] = a0; w1x[j] = a1; w1y[j] = a2; w1z[j] = a3; bb1[j] = bv;
    }
    // Block-diagonal Wc[256][16] -> swizzled B fragments (ISA 16-bit B layout:
    // lanes 0-15 hold K {0..7,16..23}, lanes 16-31 hold K {8..15,24..31}).
    for (int e = tid; e < 4096; e += 128) {
        int c = e >> 9, rem = e & 511, ln = rem >> 4, hh = rem & 15;
        int n = ln & 15, hb = ln >> 4;
        int k = c * 32 + ((hh < 8) ? (hb * 8 + hh) : (16 + hb * 8 + (hh - 8)));
        float v = 0.0f;
        if (k < 64)        { if (n == 0)           v = e_w2[k]; }
        else if (k < 128)  { if (n >= 1 && n <= 3) v = m_w2[(k - 64) * 3 + (n - 1)]; }
        else               { if (n >= 4 && n <= 9) v = s_w2[(k - 128) * 6 + (n - 4)]; }
        fragB[e] = (_Float16)v;
    }
    __syncthreads();

    const float w0 = wsf[0], w1 = wsf[1];
    const float eb2 = e_b2[0];

    const int wave = tid >> 5, lane = tid & 31;
    const int m = lane & 15, hb = lane >> 4;      // pixel row / K-half select
    _Float16* Arow = &Abuf[(wave * 16 + m) * ASTRIDE];
    const int jbase = hb * 128;                   // this lane computes 128 hiddens

    for (int it = 0; it < ITERS; ++it) {
        const int  g     = (blockIdx.x * WAVES + wave) * ITERS + it;
        const long pbase = (long)g * 16;
        const int  b     = (int)(pbase >> 16);
        const float ms = mass[pbase + m];
        const float vx = vel[b * 3 + 0], vy = vel[b * 3 + 1], vz = vel[b * 3 + 2];
        if (it + 1 < ITERS)
            __builtin_prefetch(&mass[pbase + 16 + m], 0, 0);  // global_prefetch

        // ---- layer 1 + exact GELU -> staged f16 A rows --------------------
        for (int i = 0; i < 128; i += 2) {
            int j = jbase + i;
            float p0 = fmaf(ms, w1m[j],     fmaf(vx, w1x[j],     fmaf(vy, w1y[j],     fmaf(vz, w1z[j],     bb1[j]))));
            float p1 = fmaf(ms, w1m[j + 1], fmaf(vx, w1x[j + 1], fmaf(vy, w1y[j + 1], fmaf(vz, w1z[j + 1], bb1[j + 1]))));
            v2h h2;
            h2[0] = (_Float16)gelu_exact(p0);
            h2[1] = (_Float16)gelu_exact(p1);
            *(v2h*)(Arow + j) = h2;
        }
        __syncthreads();

        // ---- second layers: 8 x v_wmma_f32_16x16x32_f16 (K = 256) ---------
        v8f acc = {0.f, 0.f, 0.f, 0.f, 0.f, 0.f, 0.f, 0.f};
#pragma unroll
        for (int c = 0; c < 8; ++c) {
            const int abase = c * 32 + hb * 8;
            v8h alo = *(const v8h*)(Arow + abase);
            v8h ahi = *(const v8h*)(Arow + abase + 16);
            v8h blo = *(const v8h*)(&fragB[(c * 32 + lane) * 16]);
            v8h bhi = *(const v8h*)(&fragB[(c * 32 + lane) * 16 + 8]);
            v16h a, bf;
#pragma unroll
            for (int q = 0; q < 8; ++q) {
                a[q] = alo[q];  a[8 + q]  = ahi[q];
                bf[q] = blo[q]; bf[8 + q] = bhi[q];
            }
            acc = __builtin_amdgcn_wmma_f32_16x16x32_f16(
                false, a, false, bf, (short)0, acc, false, false);
        }

        // ---- epilogue: lane = output channel n, 8 pixel rows --------------
        const int n = m;
        const int rowbase = hb * 8;
        if (n < 10) {
            float bias2 = (n == 0) ? eb2 : ((n <= 3) ? m_b2[n - 1] : s_b2[n - 4]);
#pragma unroll
            for (int v = 0; v < 8; ++v) {
                long p = pbase + rowbase + v;
                float t = acc[v] + bias2;
                if (n == 0) t = (t > 20.0f) ? t : log1pf(expf(t));   // softplus
                float mt = metric_init[p * 10 + n] - (w0 * COUPLING_F) * t;
                if (n == 4 || n == 7) mt = fmaf(2.0f * w1, pot[p], mt);
                if (n == 0) mt = fminf(mt, -0.1f);
                if (n == 4 || n == 7 || n == 9) mt = fmaxf(mt, 0.1f);
                metric_pre[p * 10 + n] = mt;
            }
        }
        __syncthreads();   // protect Abuf reuse next iteration
    }
}

// ---------------------------------------------------------------------------
// 3. Conditional 3x3 depthwise Gaussian smoothing (sigma=1, zero pad)
__global__ void k_smooth(const float* __restrict__ wsf,
                         const float* __restrict__ pre,
                         float* __restrict__ out, int total) {
    const bool do_smooth = wsf[2] > 0.1f;
    const float g1[3] = {0.27406862f, 0.45186276f, 0.27406862f};
    for (int idx = blockIdx.x * blockDim.x + threadIdx.x; idx < total;
         idx += gridDim.x * blockDim.x) {
        if (!do_smooth) { out[idx] = pre[idx]; continue; }
        int  ch = idx % 10;
        long p  = idx / 10;
        int  w  = (int)(p & 255);
        int  h  = (int)((p >> 8) & 255);
        long bb = p & ~65535L;
        float s = 0.0f;
#pragma unroll
        for (int dy = -1; dy <= 1; ++dy) {
            int hh = h + dy;
            if (hh < 0 || hh > 255) continue;
#pragma unroll
            for (int dx = -1; dx <= 1; ++dx) {
                int ww = w + dx;
                if (ww < 0 || ww > 255) continue;
                s = fmaf(g1[dy + 1] * g1[dx + 1],
                         pre[(bb + hh * 256 + ww) * 10 + ch], s);
            }
        }
        out[idx] = s;
    }
}

// ---------------------------------------------------------------------------
extern "C" void kernel_launch(void* const* d_in, const int* in_sizes, int n_in,
                              void* d_out, int out_size, void* d_ws, size_t ws_size,
                              hipStream_t stream) {
    const float* mass  = (const float*)d_in[0];
    const float* vel   = (const float*)d_in[1];
    const float* minit = (const float*)d_in[2];
    const float* e_w1  = (const float*)d_in[3];
    const float* e_b1  = (const float*)d_in[4];
    const float* e_w2  = (const float*)d_in[5];
    const float* e_b2  = (const float*)d_in[6];
    const float* m_w1  = (const float*)d_in[7];
    const float* m_b1  = (const float*)d_in[8];
    const float* m_w2  = (const float*)d_in[9];
    const float* m_b2  = (const float*)d_in[10];
    const float* s_w1  = (const float*)d_in[11];
    const float* s_b1  = (const float*)d_in[12];
    const float* s_w2  = (const float*)d_in[13];
    const float* s_b2  = (const float*)d_in[14];
    const float* sp    = (const float*)d_in[15];

    float* wsf = (float*)d_ws;
    float* pot = wsf + WS_POT_OFF;
    float* pre = wsf + WS_POT_OFF + NPIX;
    float* out = (float*)d_out;

    k_init<<<1, 1, 0, stream>>>(wsf);
    k_maxreduce<<<512, 256, 0, stream>>>(mass, wsf, NPIX);
    k_finalize<<<1, 1, 0, stream>>>(sp, wsf);
    k_green<<<dim3(16, 16, 8), dim3(16, 16), 0, stream>>>(mass, wsf, pot);

    // 32768 pixel-groups of 16; 4 waves x 4 iters per 128-thread block
    k_em<<<2048, 128, 0, stream>>>(mass, vel, minit,
                                   e_w1, e_b1, e_w2, e_b2,
                                   m_w1, m_b1, m_w2, m_b2,
                                   s_w1, s_b1, s_w2, s_b2,
                                   wsf, pot, pre);

    k_smooth<<<4096, 256, 0, stream>>>(wsf, pre, out, NELEM);
}
